// MiniGPT_52527450030370
// MI455X (gfx1250) — matrix operand order repaired and checked by reading, not compiled, not run on hardware
//
#include <hip/hip_runtime.h>
#include <hip/hip_bf16.h>
#include <math.h>

#define kV   32000
#define kD   256
#define kH   4
#define kDFF 1024
#define kL   4
#define kB   2
#define kT   2048
#define kM   (kB * kT)
#define kDK  64

typedef _Float16 h4   __attribute__((ext_vector_type(4)));
typedef _Float16 h8   __attribute__((ext_vector_type(8)));
typedef _Float16 v16h __attribute__((ext_vector_type(16)));
typedef float    v8f  __attribute__((ext_vector_type(8)));
typedef int      i4v  __attribute__((vector_size(16)));  // matches builtin param

#define AS1 __attribute__((address_space(1)))
#define AS3 __attribute__((address_space(3)))

// gfx1250 async global->LDS copy (ASYNCcnt) if the toolchain exposes it.
#if defined(__has_builtin)
#if __has_builtin(__builtin_amdgcn_global_load_async_to_lds_b128) && \
    __has_builtin(__builtin_amdgcn_s_wait_asynccnt)
#define USE_ASYNC_LDS 1
#endif
#endif
#ifndef USE_ASYNC_LDS
#define USE_ASYNC_LDS 0
#endif

// D = A(16x32 f16) * B(32x16 f16) + C(16x16 f32)
__device__ __forceinline__ v8f wmma_f16(v16h a, v16h b, v8f c) {
  return __builtin_amdgcn_wmma_f32_16x16x32_f16(false, a, false, b, (short)0, c,
                                                false, false);
}

// A-matrix fragment (16x32, M=lane&15, elem e <-> k=(e&7)+8*hh+16*(e>>3))
__device__ __forceinline__ v16h frag_a(const _Float16* row, int hh) {
  h8 lo = *(const h8*)(row + 8 * hh);
  h8 hi = *(const h8*)(row + 16 + 8 * hh);
  return __builtin_shufflevector(lo, hi, 0, 1, 2, 3, 4, 5, 6, 7, 8, 9, 10, 11,
                                 12, 13, 14, 15);
}
// B-matrix fragment (32x16, N=lane&15, elem e <-> k=(e&7)+8*hh+16*(e>>3))
__device__ __forceinline__ v16h frag_b(const _Float16* row, int hh) {
  h8 lo = *(const h8*)(row + 8 * hh);
  h8 hi = *(const h8*)(row + 16 + 8 * hh);
  return __builtin_shufflevector(lo, hi, 0, 1, 2, 3, 4, 5, 6, 7, 8, 9, 10, 11,
                                 12, 13, 14, 15);
}

__device__ __forceinline__ h8 pack8(float4 a, float4 b) {
  h8 r = {(_Float16)a.x, (_Float16)a.y, (_Float16)a.z, (_Float16)a.w,
          (_Float16)b.x, (_Float16)b.y, (_Float16)b.z, (_Float16)b.w};
  return r;
}

// ---------------- f32 -> f16 bulk convert (n multiple of 1024) -------------
__global__ void k_half(const float* __restrict__ src,
                       _Float16* __restrict__ dst) {
  size_t i = ((size_t)blockIdx.x * 256 + threadIdx.x) * 4;
  float4 v = *(const float4*)(src + i);
  h4 o = {(_Float16)v.x, (_Float16)v.y, (_Float16)v.z, (_Float16)v.w};
  *(h4*)(dst + i) = o;
}

// ---------------- embedding + positional encoding ----------------
__global__ void k_embed(const int* __restrict__ tokens,
                        const float* __restrict__ embed,
                        float* __restrict__ x) {
  int m = blockIdx.x;
  int d = threadIdx.x;
  int t = m % kT;
  int tok = tokens[m];
  float i2 = (float)(d & ~1);
  float freq = expf(i2 * (-9.210340371976184f / (float)kD));  // -ln(10000)/D
  float ang = (float)t * freq;
  float pe = (d & 1) ? cosf(ang) : sinf(ang);
  x[(size_t)m * kD + d] = embed[(size_t)tok * kD + d] + pe;
}

// ------- WMMA GEMM, f16 in / f32 out: C = act(A[M,K] * W[N,K]^T + bias) ----
// Double-buffered K-slices; staging via async global->LDS DMA when available.
#define GSTRIDE 40  // LDS row stride in halves (80B) -> conflict-free b128
__global__ void __launch_bounds__(128)
k_gemm(const _Float16* __restrict__ A, const _Float16* __restrict__ W,
       const float* __restrict__ bias, float* __restrict__ C,
       int Ndim, int Kdim, int relu, int ntstore) {
  __shared__ _Float16 Al[2][64 * GSTRIDE];
  __shared__ _Float16 Wl[2][64 * GSTRIDE];

  int n0 = blockIdx.x * 64;
  int m0 = blockIdx.y * 64;
  int tid = threadIdx.x;
  int lane = tid & 31, wv = tid >> 5;
  int wr = (wv >> 1) * 32;
  int wc = (wv & 1) * 32;
  int hh = lane >> 4, lc = lane & 15;

  v8f acc[2][2] = {};

  // stage one 64x32 K-slice of A and W into LDS buffer `buf`
  auto stage = [&](int buf, int kk) {
#pragma unroll
    for (int i = 0; i < 2; i++) {
      int j = tid + 128 * i;        // 0..255
      int r = j >> 2, seg = j & 3;  // 64 rows x 4 segments of 8 halves
      const _Float16* ga = &A[(size_t)(m0 + r) * Kdim + kk + seg * 8];
      const _Float16* gw = &W[(size_t)(n0 + r) * Kdim + kk + seg * 8];
      _Float16* la = &Al[buf][r * GSTRIDE + seg * 8];
      _Float16* lw = &Wl[buf][r * GSTRIDE + seg * 8];
#if USE_ASYNC_LDS
      __builtin_amdgcn_global_load_async_to_lds_b128((AS1 i4v*)ga, (AS3 i4v*)la,
                                                     0, 0);
      __builtin_amdgcn_global_load_async_to_lds_b128((AS1 i4v*)gw, (AS3 i4v*)lw,
                                                     0, 0);
#else
      *(h8*)la = *(const h8*)ga;
      *(h8*)lw = *(const h8*)gw;
#endif
    }
  };

  int nk = Kdim / 32;
  stage(0, 0);
  for (int t = 0; t < nk; t++) {
    int cur = t & 1;
#if USE_ASYNC_LDS
    __builtin_amdgcn_s_wait_asynccnt(0);  // my async loads into buf[cur] done
#endif
    __syncthreads();  // all waves' stores/loads into buf[cur] visible
    if (t + 1 < nk) stage(cur ^ 1, (t + 1) * 32);  // prefetch next slice

    const _Float16* Ab = Al[cur];
    const _Float16* Wb = Wl[cur];
    v16h af0 = frag_a(&Ab[(wr + lc) * GSTRIDE], hh);
    v16h af1 = frag_a(&Ab[(wr + 16 + lc) * GSTRIDE], hh);
    v16h bf0 = frag_b(&Wb[(wc + lc) * GSTRIDE], hh);
    v16h bf1 = frag_b(&Wb[(wc + 16 + lc) * GSTRIDE], hh);

    acc[0][0] = wmma_f16(af0, bf0, acc[0][0]);
    acc[0][1] = wmma_f16(af0, bf1, acc[0][1]);
    acc[1][0] = wmma_f16(af1, bf0, acc[1][0]);
    acc[1][1] = wmma_f16(af1, bf1, acc[1][1]);
  }

#pragma unroll
  for (int j = 0; j < 2; j++) {
    int gc = n0 + wc + j * 16 + lc;
    float bv = bias[gc];
#pragma unroll
    for (int i = 0; i < 2; i++) {
      int grb = m0 + wr + i * 16 + 8 * hh;
#pragma unroll
      for (int v = 0; v < 8; v++) {
        float val = acc[i][j][v] + bv;
        if (relu) val = fmaxf(val, 0.f);
        float* dst = &C[(size_t)(grb + v) * Ndim + gc];
        if (ntstore)
          __builtin_nontemporal_store(val, dst);
        else
          *dst = val;
      }
    }
  }
}

// ---------------- flash attention (causal), 2 waves = 32 q-rows / block -----
#define QSTRIDE 72  // 64-wide rows padded (144B) -> conflict-free b128 reads
__global__ void __launch_bounds__(64)
k_attn(const float* __restrict__ qkv, float* __restrict__ ob) {
  __shared__ _Float16 Ql[32 * QSTRIDE];
  __shared__ _Float16 Kl[32 * QSTRIDE];
  __shared__ _Float16 Vt[64 * GSTRIDE];
  __shared__ float    Sl[2][16 * 33];
  __shared__ _Float16 Pl[2][16 * GSTRIDE];
  __shared__ float    Aw[2][16];
  __shared__ float    Lw[2][16];

  int q0 = blockIdx.x * 32;
  int h = blockIdx.y;
  int b = blockIdx.z;
  int tid = threadIdx.x, lane = tid & 31, wv = tid >> 5;
  int hh = lane >> 4, lc = lane & 15;
  size_t base = (size_t)b * kT * (3 * kD);

#pragma unroll
  for (int i = 0; i < 4; i++) {
    int j = tid + 64 * i;
    int r = j >> 3, seg = j & 7;
    const float4* p = (const float4*)&qkv[base + (size_t)(q0 + r) * (3 * kD) +
                                          h * kDK + seg * 8];
    *(h8*)&Ql[r * QSTRIDE + seg * 8] = pack8(p[0], p[1]);
  }
  __syncthreads();
  v16h qf0 = frag_a(&Ql[(wv * 16 + lc) * QSTRIDE + 0], hh);
  v16h qf1 = frag_a(&Ql[(wv * 16 + lc) * QSTRIDE + 32], hh);

  v8f ov[4] = {};
  float m_row = -INFINITY, l_row = 0.f;
  const float scale = 0.125f;  // 1/sqrt(64)

  for (int j0 = 0; j0 <= q0; j0 += 32) {
    __syncthreads();
#pragma unroll
    for (int i = 0; i < 4; i++) {
      int j = tid + 64 * i;
      int r = j >> 3, seg = j & 7;
      const float4* p = (const float4*)&qkv[base + (size_t)(j0 + r) * (3 * kD) +
                                            kD + h * kDK + seg * 8];
      *(h8*)&Kl[r * QSTRIDE + seg * 8] = pack8(p[0], p[1]);
    }
    {  // V^T: thread = dk column, pack 4 keys -> ds_store_b64
      int c = tid;
      const float* vcol =
          &qkv[base + (size_t)j0 * (3 * kD) + 2 * kD + h * kDK + c];
#pragma unroll
      for (int r0 = 0; r0 < 32; r0 += 4) {
        float a0 = vcol[(size_t)(r0 + 0) * (3 * kD)];
        float a1 = vcol[(size_t)(r0 + 1) * (3 * kD)];
        float a2 = vcol[(size_t)(r0 + 2) * (3 * kD)];
        float a3 = vcol[(size_t)(r0 + 3) * (3 * kD)];
        h4 pk = {(_Float16)a0, (_Float16)a1, (_Float16)a2, (_Float16)a3};
        *(h4*)&Vt[c * GSTRIDE + r0] = pk;
      }
    }
    __syncthreads();

    v8f s0 = {}, s1 = {};
    s0 = wmma_f16(qf0, frag_b(&Kl[lc * QSTRIDE + 0], hh), s0);
    s0 = wmma_f16(qf1, frag_b(&Kl[lc * QSTRIDE + 32], hh), s0);
    s1 = wmma_f16(qf0, frag_b(&Kl[(16 + lc) * QSTRIDE + 0], hh), s1);
    s1 = wmma_f16(qf1, frag_b(&Kl[(16 + lc) * QSTRIDE + 32], hh), s1);
#pragma unroll
    for (int v = 0; v < 8; v++) {
      Sl[wv][(v + 8 * hh) * 33 + lc] = s0[v];
      Sl[wv][(v + 8 * hh) * 33 + 16 + lc] = s1[v];
    }
    __syncthreads();

    if (lane < 16) {
      int r = lane;
      int q = q0 + wv * 16 + r;
      float sv[32];
      float mx = m_row;
#pragma unroll
      for (int j = 0; j < 32; j++) {
        float s = Sl[wv][r * 33 + j] * scale;
        if (j0 + j > q) s = -INFINITY;
        sv[j] = s;
        mx = fmaxf(mx, s);
      }
      float alpha = __expf(m_row - mx);
      float ls = l_row * alpha;
#pragma unroll
      for (int j = 0; j < 32; j++) {
        float p = __expf(sv[j] - mx);
        ls += p;
        Pl[wv][r * GSTRIDE + j] = (_Float16)p;
      }
      m_row = mx;
      l_row = ls;
      Aw[wv][r] = alpha;
    }
    __syncthreads();

    float av[8];
#pragma unroll
    for (int v = 0; v < 8; v++) av[v] = Aw[wv][v + 8 * hh];
#pragma unroll
    for (int t = 0; t < 4; t++)
#pragma unroll
      for (int v = 0; v < 8; v++) ov[t][v] *= av[v];

    v16h pf = frag_a(&Pl[wv][lc * GSTRIDE], hh);
#pragma unroll
    for (int nt = 0; nt < 4; nt++)
      ov[nt] = wmma_f16(pf, frag_b(&Vt[(nt * 16 + lc) * GSTRIDE], hh), ov[nt]);
  }

  if (lane < 16) Lw[wv][lane] = l_row;
  __syncthreads();
  float inv[8];
#pragma unroll
  for (int v = 0; v < 8; v++) inv[v] = 1.f / Lw[wv][v + 8 * hh];
#pragma unroll
  for (int nt = 0; nt < 4; nt++)
#pragma unroll
    for (int v = 0; v < 8; v++) {
      int q = q0 + wv * 16 + v + 8 * hh;
      ob[((size_t)b * kT + q) * kD + h * kDK + nt * 16 + lc] = ov[nt][v] * inv[v];
    }
}

// ---------------- residual add + LayerNorm (in-place on x) ------------------
__global__ void __launch_bounds__(256)
k_add_ln(float* __restrict__ x, const float* __restrict__ res,
         const float* __restrict__ s, const float* __restrict__ bta,
         int has_res) {
  __shared__ float red[kD];
  int row = blockIdx.x, d = threadIdx.x;
  size_t idx = (size_t)row * kD + d;
  float y = x[idx] + (has_res ? res[idx] : 0.f);
  red[d] = y;
  __syncthreads();
  for (int off = kD / 2; off > 0; off >>= 1) {
    if (d < off) red[d] += red[d + off];
    __syncthreads();
  }
  float mu = red[0] * (1.f / kD);
  __syncthreads();
  float c = y - mu;
  red[d] = c * c;
  __syncthreads();
  for (int off = kD / 2; off > 0; off >>= 1) {
    if (d < off) red[d] += red[d + off];
    __syncthreads();
  }
  float var = red[0] * (1.f / kD);
  x[idx] = c * rsqrtf(var + 1e-5f) * s[d] + bta[d];
}

extern "C" void kernel_launch(void* const* d_in, const int* in_sizes, int n_in,
                              void* d_out, int out_size, void* d_ws,
                              size_t ws_size, hipStream_t stream) {
  (void)in_sizes; (void)n_in; (void)out_size; (void)ws_size;
  const int*   tokens = (const int*)d_in[0];
  const float* embed  = (const float*)d_in[1];
  const float* qkv_w  = (const float*)d_in[2];
  const float* qkv_b  = (const float*)d_in[3];
  const float* fc_w   = (const float*)d_in[4];
  const float* fc_b   = (const float*)d_in[5];
  const float* ln1_s  = (const float*)d_in[6];
  const float* ln1_b  = (const float*)d_in[7];
  const float* w1     = (const float*)d_in[8];
  const float* b1     = (const float*)d_in[9];
  const float* w2     = (const float*)d_in[10];
  const float* b2     = (const float*)d_in[11];
  const float* ln2_s  = (const float*)d_in[12];
  const float* ln2_b  = (const float*)d_in[13];
  const float* lnf_s  = (const float*)d_in[14];
  const float* lnf_b  = (const float*)d_in[15];
  const float* out_w  = (const float*)d_in[16];
  const float* out_b  = (const float*)d_in[17];

  // workspace layout
  float* xs  = (float*)d_ws;                       // [M,D]   f32   4MB
  float* big = xs + (size_t)kM * kD;               // [M,DFF] f32  16MB
  float* ob  = big + (size_t)kM * kDFF;            // [M,D]   f32   4MB
  _Float16* a16    = (_Float16*)(ob + (size_t)kM * kD);  // [M,DFF] f16 8MB
  _Float16* qkvw16 = a16 + (size_t)kM * kDFF;
  _Float16* fcw16  = qkvw16 + (size_t)kL * 3 * kD * kD;
  _Float16* w1h    = fcw16 + (size_t)kL * kD * kD;
  _Float16* w2h    = w1h + (size_t)kL * kDFF * kD;
  _Float16* owh    = w2h + (size_t)kL * kD * kDFF;  // [V,D] f16 16MB

  // one-time (per launch) weight conversion to f16
  k_half<<<dim3((kL * 3 * kD * kD) / 1024), dim3(256), 0, stream>>>(qkv_w, qkvw16);
  k_half<<<dim3((kL * kD * kD) / 1024), dim3(256), 0, stream>>>(fc_w, fcw16);
  k_half<<<dim3((kL * kDFF * kD) / 1024), dim3(256), 0, stream>>>(w1, w1h);
  k_half<<<dim3((kL * kD * kDFF) / 1024), dim3(256), 0, stream>>>(w2, w2h);
  k_half<<<dim3((kV * kD) / 1024), dim3(256), 0, stream>>>(out_w, owh);

  k_embed<<<dim3(kM), dim3(kD), 0, stream>>>(tokens, embed, xs);

  for (int l = 0; l < kL; l++) {
    // qkv = x @ qkv_w^T + qkv_b      [M, 768]
    k_half<<<dim3((kM * kD) / 1024), dim3(256), 0, stream>>>(xs, a16);
    k_gemm<<<dim3((3 * kD) / 64, kM / 64), dim3(128), 0, stream>>>(
        a16, qkvw16 + (size_t)l * 3 * kD * kD, qkv_b + (size_t)l * 3 * kD, big,
        3 * kD, kD, 0, 0);
    // o = attention(qkv)             [M, 256]
    k_attn<<<dim3(kT / 32, kH, kB), dim3(64), 0, stream>>>(big, ob);
    // proj = o @ fc_w^T + fc_b       [M, 256]  (reuse big)
    k_half<<<dim3((kM * kD) / 1024), dim3(256), 0, stream>>>(ob, a16);
    k_gemm<<<dim3(kD / 64, kM / 64), dim3(128), 0, stream>>>(
        a16, fcw16 + (size_t)l * kD * kD, fc_b + (size_t)l * kD, big, kD, kD, 0,
        0);
    // x = LN(x + proj)
    k_add_ln<<<dim3(kM), dim3(kD), 0, stream>>>(
        xs, big, ln1_s + (size_t)l * kD, ln1_b + (size_t)l * kD, 1);
    // h = relu(x @ w1^T + b1)        [M, 1024]
    k_half<<<dim3((kM * kD) / 1024), dim3(256), 0, stream>>>(xs, a16);
    k_gemm<<<dim3(kDFF / 64, kM / 64), dim3(128), 0, stream>>>(
        a16, w1h + (size_t)l * kDFF * kD, b1 + (size_t)l * kDFF, big, kDFF, kD,
        1, 0);
    // f = h @ w2^T + b2              [M, 256]
    k_half<<<dim3((kM * kDFF) / 1024), dim3(256), 0, stream>>>(big, a16);
    k_gemm<<<dim3(kD / 64, kM / 64), dim3(128), 0, stream>>>(
        a16, w2h + (size_t)l * kD * kDFF, b2 + (size_t)l * kD, ob, kD, kDFF, 0,
        0);
    // x = LN(x + f)
    k_add_ln<<<dim3(kM), dim3(kD), 0, stream>>>(
        xs, ob, ln2_s + (size_t)l * kD, ln2_b + (size_t)l * kD, 1);
  }

  // final LN, then logits = x @ out_w^T + out_b  -> d_out
  k_add_ln<<<dim3(kM), dim3(kD), 0, stream>>>(xs, ob, lnf_s, lnf_b, 0);
  k_half<<<dim3((kM * kD) / 1024), dim3(256), 0, stream>>>(xs, a16);
  // 524MB write-once stream: non-temporal stores so it doesn't evict L2.
  k_gemm<<<dim3(kV / 64, kM / 64), dim3(128), 0, stream>>>(
      a16, owh, out_b, (float*)d_out, kV, kD, 0, 1);
}
